// HET_EglRelGraphConv_9998683865829
// MI455X (gfx1250) — matrix operand order repaired and checked by reading, not compiled, not run on hardware
//
#include <hip/hip_runtime.h>

typedef __attribute__((ext_vector_type(16))) __bf16 v16bf;
typedef __attribute__((ext_vector_type(8)))  float  v8f;

#define IN_FEAT  64
#define OUT_FEAT 64
#define NUM_RELS 8
#define NUM_BASES 4
#define CHUNK    2048
#define EPT      (CHUNK / 256)   // edges per thread in the local sort

struct TailT  { static constexpr bool value = true;  };
struct FullT  { static constexpr bool value = false; };

// float -> bf16 (let the compiler pick the native gfx1250 conversion)
__device__ __forceinline__ __bf16 f2bf(float f) { return (__bf16)f; }

__device__ __forceinline__ void pack8(v16bf& a, int at, float4 lo, float4 hi) {
  a[at + 0] = f2bf(lo.x); a[at + 1] = f2bf(lo.y);
  a[at + 2] = f2bf(lo.z); a[at + 3] = f2bf(lo.w);
  a[at + 4] = f2bf(hi.x); a[at + 5] = f2bf(hi.y);
  a[at + 6] = f2bf(hi.z); a[at + 7] = f2bf(hi.w);
}

// rel_w[r][o][i] = sum_b w_comp[r,b] * weight[b,i,o]  (transposed, bf16)
__global__ void build_relw_kernel(const float* __restrict__ weight,
                                  const float* __restrict__ w_comp,
                                  __bf16* __restrict__ relw) {
  int idx = blockIdx.x * blockDim.x + threadIdx.x;
  const int total = NUM_RELS * OUT_FEAT * IN_FEAT;
  for (; idx < total; idx += gridDim.x * blockDim.x) {
    int r = idx >> 12;
    int o = (idx >> 6) & 63;
    int i = idx & 63;
    float acc = 0.f;
#pragma unroll
    for (int b = 0; b < NUM_BASES; ++b)
      acc += w_comp[r * NUM_BASES + b] * weight[(b * IN_FEAT + i) * OUT_FEAT + o];
    relw[idx] = f2bf(acc);
  }
}

__global__ void __launch_bounds__(256)
rgcn_edge_wmma_kernel(const float* __restrict__ x,
                      const __bf16* __restrict__ relw,
                      const float* __restrict__ norm,
                      const int* __restrict__ src,
                      const int* __restrict__ dst,
                      const int* __restrict__ etypes,
                      float* __restrict__ out,
                      int nEdges) {
  __shared__ int s_idx[CHUNK];
  __shared__ int s_cnt[NUM_RELS];
  __shared__ int s_off[NUM_RELS];

  const int tid  = threadIdx.x;
  const int base = blockIdx.x * CHUNK;

  if (tid < NUM_RELS) s_cnt[tid] = 0;
  __syncthreads();

  // ---- block-local counting sort of edge indices by relation ----
  int myRank[EPT], myEdge[EPT], myEt[EPT];
#pragma unroll
  for (int j = 0; j < EPT; ++j) {
    int e = base + j * 256 + tid;
    if (e < nEdges) {
      int et = etypes[e];
      myEt[j] = et;
      myEdge[j] = e;
      myRank[j] = atomicAdd(&s_cnt[et], 1);
    } else {
      myEt[j] = -1;
    }
  }
  __syncthreads();
  if (tid == 0) {
    int o = 0;
#pragma unroll
    for (int r = 0; r < NUM_RELS; ++r) { s_off[r] = o; o += s_cnt[r]; }
  }
  __syncthreads();
#pragma unroll
  for (int j = 0; j < EPT; ++j)
    if (myEt[j] >= 0) s_idx[s_off[myEt[j]] + myRank[j]] = myEdge[j];
  __syncthreads();

  // ---- wave w handles relation w for this block's edges ----
  const int wave = tid >> 5;
  const int lane = tid & 31;
  const int cnt  = s_cnt[wave];
  const int off  = s_off[wave];
  if (cnt == 0) return;

  const int ncol = lane & 15;   // N for B/C tiles, M for A tile
  const int kh   = lane >> 4;   // lane-half selects K sub-range

  // B fragments for this wave's relation: 2 K-blocks x 4 N-tiles,
  // hoisted into VGPRs for the whole group loop.
  // relw stored [r][n][k] bf16 -> each fragment = 16 contiguous halves (32 B).
  v16bf B[2][4];
#pragma unroll
  for (int kb = 0; kb < 2; ++kb)
#pragma unroll
    for (int nt = 0; nt < 4; ++nt) {
      int n = nt * 16 + ncol;
      B[kb][nt] = *(const v16bf*)(relw +
                   (unsigned)((wave * 64 + n) * 64 + kb * 32 + kh * 16));
    }

  const int mbase = kh * 8;     // C VGPR v -> row M = mbase + v

  auto group_body = [&](int g, auto tailc) {
    constexpr bool TAIL = decltype(tailc)::value;
    const int seg = g * 16 + ncol;
    const bool mvalid = !TAIL || (seg < cnt);
    const int eidx = s_idx[off + (TAIL ? (mvalid ? seg : 0) : seg)];
    const float nrm = norm[eidx];
    const int   srow = src[eidx];
    const int   drow = dst[eidx];

    // Gather src row of x (f32, L2-resident) -> bf16 A fragments.
    // A k order chosen here (matches B): lane half h -> K base 16h; v0..7 = K base+0..15
    // (per 16x32 tile).
    const float* xp = x + (unsigned)(srow * IN_FEAT + kh * 16);
    float4 q0 = ((const float4*)xp)[0];
    float4 q1 = ((const float4*)xp)[1];
    float4 q2 = ((const float4*)(xp + 8))[0];
    float4 q3 = ((const float4*)(xp + 8))[1];
    float4 q4 = ((const float4*)(xp + 32))[0];
    float4 q5 = ((const float4*)(xp + 32))[1];
    float4 q6 = ((const float4*)(xp + 40))[0];
    float4 q7 = ((const float4*)(xp + 40))[1];
    v16bf A0, A1;
    pack8(A0, 0, q0, q1);
    pack8(A0, 8, q2, q3);
    pack8(A1, 0, q4, q5);
    pack8(A1, 8, q6, q7);

    // All four 16-wide output tiles (K accumulated over both 32-blocks).
    v8f c0 = {}, c1 = {}, c2 = {}, c3 = {};
    c0 = __builtin_amdgcn_wmma_f32_16x16x32_bf16(false, A0, false, B[0][0], (short)0, c0, false, false);
    c0 = __builtin_amdgcn_wmma_f32_16x16x32_bf16(false, A1, false, B[1][0], (short)0, c0, false, false);
    c1 = __builtin_amdgcn_wmma_f32_16x16x32_bf16(false, A0, false, B[0][1], (short)0, c1, false, false);
    c1 = __builtin_amdgcn_wmma_f32_16x16x32_bf16(false, A1, false, B[1][1], (short)0, c1, false, false);
    c2 = __builtin_amdgcn_wmma_f32_16x16x32_bf16(false, A0, false, B[0][2], (short)0, c2, false, false);
    c2 = __builtin_amdgcn_wmma_f32_16x16x32_bf16(false, A1, false, B[1][2], (short)0, c2, false, false);
    c3 = __builtin_amdgcn_wmma_f32_16x16x32_bf16(false, A0, false, B[0][3], (short)0, c3, false, false);
    c3 = __builtin_amdgcn_wmma_f32_16x16x32_bf16(false, A1, false, B[1][3], (short)0, c3, false, false);

    // Scatter: C/D layout lane -> N = lane&15, VGPR v -> M = mbase + v.
    // Apply norm to the f32 result (more accurate than folding into bf16 A).
    // One 32-bit offset per row; N-tiles reached via immediate offsets.
#pragma unroll
    for (int v = 0; v < 8; ++v) {
      const int   d  = __shfl(drow, mbase + v, 32);
      const float nv = __shfl(nrm,  mbase + v, 32);
      const bool ok = !TAIL || ((g * 16 + mbase + v) < cnt);
      if (ok) {
        const unsigned o0 = (unsigned)(d * OUT_FEAT + ncol);
        atomicAdd(out + o0,      c0[v] * nv);
        atomicAdd(out + o0 + 16, c1[v] * nv);
        atomicAdd(out + o0 + 32, c2[v] * nv);
        atomicAdd(out + o0 + 48, c3[v] * nv);
      }
    }
  };

  const int nfull   = cnt >> 4;
  const int ngroups = (cnt + 15) >> 4;
  for (int g = 0; g < nfull; ++g) group_body(g, FullT{});
  if (ngroups > nfull)            group_body(nfull, TailT{});
}

__global__ void bias_relu_kernel(float* __restrict__ out,
                                 const float* __restrict__ h_bias, int total) {
  int i = blockIdx.x * blockDim.x + threadIdx.x;
  if (i < total) {
    float v = out[i] + h_bias[i & (OUT_FEAT - 1)];
    out[i] = v > 0.f ? v : 0.f;
  }
}

extern "C" void kernel_launch(void* const* d_in, const int* in_sizes, int n_in,
                              void* d_out, int out_size, void* d_ws, size_t ws_size,
                              hipStream_t stream) {
  const float* x      = (const float*)d_in[0];
  const float* weight = (const float*)d_in[1];
  const float* w_comp = (const float*)d_in[2];
  const float* h_bias = (const float*)d_in[3];
  const float* norm   = (const float*)d_in[4];
  const int*   src    = (const int*)d_in[5];
  const int*   dst    = (const int*)d_in[6];
  const int*   et     = (const int*)d_in[7];

  const int nEdges = in_sizes[5];
  const int nNodes = in_sizes[0] / IN_FEAT;

  float*  out  = (float*)d_out;
  __bf16* relw = (__bf16*)d_ws;   // 8*64*64*2 = 64 KB of scratch

  hipMemsetAsync(out, 0, (size_t)nNodes * OUT_FEAT * sizeof(float), stream);
  build_relw_kernel<<<8, 256, 0, stream>>>(weight, w_comp, relw);

  const int nblk = (nEdges + CHUNK - 1) / CHUNK;
  rgcn_edge_wmma_kernel<<<nblk, 256, 0, stream>>>(x, relw, norm, src, dst, et,
                                                  out, nEdges);

  const int total = nNodes * OUT_FEAT;
  bias_relu_kernel<<<(total + 255) / 256, 256, 0, stream>>>(out, h_bias, total);
}
